// MultiheadAttentionWithRelativePosition_23373212024842
// MI455X (gfx1250) — compile-verified, matrix operand order checked by reading, not run on hardware
//
#include <hip/hip_runtime.h>
#include <hip/hip_bf16.h>

typedef __attribute__((ext_vector_type(16))) _Float16 v16h;
typedef __attribute__((ext_vector_type(8)))  _Float16 v8h;
typedef __attribute__((ext_vector_type(8)))  float    v8f;

// Problem constants
constexpr int Bc = 8, Sc = 1024, Ec = 1024, Hc = 16, Dc = 64;
constexpr int MAXREL = 32;
constexpr int NREL = 2 * MAXREL + 1;   // 65
constexpr float SCALE = 0.125f;        // 1/sqrt(D)

__device__ __forceinline__ v8f wmma16(v16h a, v16h b, v8f c) {
  // D = A(16x32 f16) * B(32x16 f16) + C(16x16 f32)
  return __builtin_amdgcn_wmma_f32_16x16x32_f16(false, a, false, b, (short)0, c,
                                                false, false);
}

// load 16 contiguous f32 (LDS or global), convert to packed v16h fragment
__device__ __forceinline__ v16h cvt16f(const float* p) {
  const float4* q4 = (const float4*)p;
  float4 a = q4[0], b = q4[1], c = q4[2], d = q4[3];
  v16h r;
  r[0]=(_Float16)a.x; r[1]=(_Float16)a.y; r[2]=(_Float16)a.z; r[3]=(_Float16)a.w;
  r[4]=(_Float16)b.x; r[5]=(_Float16)b.y; r[6]=(_Float16)b.z; r[7]=(_Float16)b.w;
  r[8]=(_Float16)c.x; r[9]=(_Float16)c.y; r[10]=(_Float16)c.z; r[11]=(_Float16)c.w;
  r[12]=(_Float16)d.x; r[13]=(_Float16)d.y; r[14]=(_Float16)d.z; r[15]=(_Float16)d.w;
  return r;
}

__device__ __forceinline__ v16h zero16h() {
  v16h z;
  #pragma unroll
  for (int i = 0; i < 16; ++i) z[i] = (_Float16)0.0f;
  return z;
}

// ---------------------------------------------------------------------------
// One-shot f32 -> f16 conversion (8 elements / thread, b128 in / b128 out)
// ---------------------------------------------------------------------------
__global__ void f32_to_f16_kernel(const float* __restrict__ src,
                                  _Float16* __restrict__ dst, int n8) {
  const int i = blockIdx.x * blockDim.x + threadIdx.x;
  if (i >= n8) return;
  const float4* s4 = (const float4*)src;
  float4 a = s4[2 * i], b = s4[2 * i + 1];
  v8h r;
  r[0]=(_Float16)a.x; r[1]=(_Float16)a.y; r[2]=(_Float16)a.z; r[3]=(_Float16)a.w;
  r[4]=(_Float16)b.x; r[5]=(_Float16)b.y; r[6]=(_Float16)b.z; r[7]=(_Float16)b.w;
  ((v8h*)dst)[i] = r;
}

// ---------------------------------------------------------------------------
// Generic GEMM on f16:  C[m,n] = sum_k A[m,k] * W[n,k] + bias[n]
// M = B*S = 8192, N = K = E = 1024.  Pure b128 + WMMA inner loop.
// mode 0/1: Q/K -> f16 [((b*H+h)*S+s)*D+d]
// mode 2:   V   -> f16 [((b*H+h)*D+d)*S+s]  (transposed for P@V B-fragments)
// mode 3:   O   -> f32 out[m*E+n]
// Block: 128 threads (4 waves); wave tile 16x64; block tile 64x64.
// ---------------------------------------------------------------------------
__global__ void gemm_f16_kernel(const _Float16* __restrict__ A,
                                const _Float16* __restrict__ W,
                                const float* __restrict__ bias,
                                _Float16* __restrict__ out16,
                                float* __restrict__ out32,
                                int mode) {
  const int lane = threadIdx.x & 31;
  const int wave = threadIdx.x >> 5;
  const int m0 = blockIdx.x * 64 + wave * 16;
  const int n0 = blockIdx.y * 64;
  const int hi = lane >> 4;          // 0/1 : which 16-wide K slab
  const int lo = lane & 15;

  v8f acc[4] = {};

  const _Float16* arow = A + (size_t)(m0 + lo) * Ec;
  for (int kk = 0; kk < Ec; kk += 32) {
    const int ko = kk + (hi << 4);
    v16h a = *(const v16h*)(arow + ko);
    if (kk + 32 < Ec) __builtin_prefetch(arow + ko + 32, 0, 0);
    #pragma unroll
    for (int nt = 0; nt < 4; ++nt) {
      v16h b = *(const v16h*)(W + (size_t)(n0 + nt * 16 + lo) * Ec + ko);
      acc[nt] = wmma16(a, b, acc[nt]);
    }
  }

  #pragma unroll
  for (int nt = 0; nt < 4; ++nt) {
    #pragma unroll
    for (int r = 0; r < 8; ++r) {
      const int row = m0 + r + 8 * hi;       // m index
      const int col = n0 + nt * 16 + lo;     // n index
      const float val = acc[nt][r] + bias[col];
      const int bb = row >> 10, ss = row & 1023;
      const int hh = col >> 6,  dd = col & 63;
      if (mode == 0 || mode == 1) {
        out16[(((size_t)(bb * Hc + hh)) * Sc + ss) * Dc + dd] = (_Float16)val;
      } else if (mode == 2) {
        out16[(((size_t)(bb * Hc + hh)) * Dc + dd) * Sc + ss] = (_Float16)val;
      } else {
        out32[(size_t)row * Ec + col] = val;
      }
    }
  }
}

// ---------------------------------------------------------------------------
// Relative-position scores:  R[m, j] = sum_d qp_flat[m, d] * table16[j, d]
// m in [0, B*H*S), j in [0, 65).  K = D = 64 (two WMMA k-steps).
// ---------------------------------------------------------------------------
__global__ void rel_scores_kernel(const _Float16* __restrict__ qp,
                                  const _Float16* __restrict__ table16,
                                  float* __restrict__ rel) {
  const int lane = threadIdx.x & 31;
  const int wave = threadIdx.x >> 5;
  const int m0 = blockIdx.x * 64 + wave * 16;
  const int hi = lane >> 4, lo = lane & 15;

  const _Float16* qrow = qp + (size_t)(m0 + lo) * Dc;
  v16h a0 = *(const v16h*)(qrow + (hi << 4));
  v16h a1 = *(const v16h*)(qrow + 32 + (hi << 4));

  #pragma unroll
  for (int nt = 0; nt < 5; ++nt) {
    const int col = nt * 16 + lo;
    v16h b0, b1;
    if (col < NREL) {
      b0 = *(const v16h*)(table16 + (size_t)col * Dc + (hi << 4));
      b1 = *(const v16h*)(table16 + (size_t)col * Dc + 32 + (hi << 4));
    } else {
      b0 = zero16h(); b1 = zero16h();
    }
    v8f c = {};
    c = wmma16(a0, b0, c);
    c = wmma16(a1, b1, c);
    #pragma unroll
    for (int r = 0; r < 8; ++r) {
      if (col < NREL)
        rel[(size_t)(m0 + r + 8 * hi) * NREL + col] = c[r];
    }
  }
}

// ---------------------------------------------------------------------------
// Async staging of one contiguous 2 KB K-tile (16 rows x 64 d, f16) into LDS
// via GLOBAL_LOAD_ASYNC_TO_LDS_B128 (ASYNCcnt-tracked, 4 b128 ops per wave).
// ---------------------------------------------------------------------------
__device__ __forceinline__ void stage_ktile(const _Float16* kp_bh, int kt,
                                            _Float16* dstbuf, int lane) {
  const char* g = (const char*)(kp_bh + (size_t)kt * 1024);  // 2048B tile
  const unsigned lds = (unsigned)(uintptr_t)dstbuf;          // LDS byte offset
  #pragma unroll
  for (int i = 0; i < 4; ++i) {
    unsigned loff = lds + (unsigned)((i * 32 + lane) * 16);
    const char* ga = g + (i * 32 + lane) * 16;
    asm volatile("global_load_async_to_lds_b128 %0, %1, off"
                 :: "v"(loff), "v"(ga) : "memory");
  }
}

// ---------------------------------------------------------------------------
// Fused attention: scores (QK^T + rel, mask, scale) -> LDS strip -> softmax
// -> write attn_weights -> P@V -> f16 head-concat output.
// Grid: (S/32, B*H). Block: 64 threads = 2 waves; each wave owns 16 q-rows.
// Dynamic LDS: 32*1024*4B scores + 2 waves * 2 * 2KB K staging = 136 KB.
// ---------------------------------------------------------------------------
__global__ void attn_kernel(const _Float16* __restrict__ qp,
                            const _Float16* __restrict__ kp,
                            const _Float16* __restrict__ vpT,
                            const float* __restrict__ rel,
                            const unsigned char* __restrict__ mask,
                            float* __restrict__ attn_out,
                            _Float16* __restrict__ out_h) {
  extern __shared__ char smem_raw[];
  float* sc = (float*)smem_raw;                               // [32][1024] f32
  const int lane = threadIdx.x & 31;
  const int wave = threadIdx.x >> 5;
  _Float16* kstage =
      (_Float16*)(smem_raw + 32 * Sc * sizeof(float)) + wave * 2048;
  const int hi = lane >> 4, lo = lane & 15;
  const int bh = blockIdx.y;
  const int b = bh >> 4, h = bh & 15;
  const int q0 = blockIdx.x * 32 + wave * 16; // first q row of this wave

  // --- Phase 1: scores -> LDS (K tiles double-buffered via async copy) ---
  const _Float16* qrow = qp + (size_t)(bh * Sc + q0 + lo) * Dc;
  v16h aq0 = *(const v16h*)(qrow + (hi << 4));
  v16h aq1 = *(const v16h*)(qrow + 32 + (hi << 4));

  const _Float16* kp_bh = kp + (size_t)bh * Sc * Dc;
  stage_ktile(kp_bh, 0, kstage, lane);

  for (int kt = 0; kt < Sc / 16; ++kt) {
    const _Float16* cur = kstage + (kt & 1) * 1024;
    if (kt + 1 < Sc / 16) {
      // ensure prior reads of the buffer we are about to overwrite retired
      asm volatile("s_wait_dscnt 0x0" ::: "memory");
      stage_ktile(kp_bh, kt + 1, kstage + ((kt + 1) & 1) * 1024, lane);
      // async loads complete in order: <=4 outstanding => tile kt landed
      asm volatile("s_wait_asynccnt 0x4" ::: "memory");
    } else {
      asm volatile("s_wait_asynccnt 0x0" ::: "memory");
    }

    const int kc = kt * 16 + lo;              // this lane's key column
    v16h bk0 = *(const v16h*)(cur + lo * Dc + (hi << 4));
    v16h bk1 = *(const v16h*)(cur + lo * Dc + 32 + (hi << 4));

    v8f c;
    #pragma unroll
    for (int r = 0; r < 8; ++r) {
      const int qr = q0 + r + 8 * hi;
      int dlt = qr - kc;
      dlt = (dlt < -MAXREL) ? -MAXREL : (dlt > MAXREL ? MAXREL : dlt);
      c[r] = rel[(size_t)(bh * Sc + qr) * NREL + (dlt + MAXREL)];
    }
    c = wmma16(aq0, bk0, c);
    c = wmma16(aq1, bk1, c);

    const bool msk = mask[b * Sc + kc] != 0;
    #pragma unroll
    for (int r = 0; r < 8; ++r) {
      const float v = msk ? -3.0e38f : c[r] * SCALE;
      sc[(wave * 16 + r + 8 * hi) * Sc + kc] = v;
    }
  }
  __syncthreads();

  // --- Phase 2: softmax per row; write attn_weights ---------------------
  for (int r = 0; r < 16; ++r) {
    float* row = sc + (wave * 16 + r) * Sc;
    float m = -3.4e38f;
    for (int i = lane; i < Sc; i += 32) m = fmaxf(m, row[i]);
    #pragma unroll
    for (int off = 16; off >= 1; off >>= 1) m = fmaxf(m, __shfl_xor(m, off, 32));
    float ssum = 0.0f;
    for (int i = lane; i < Sc; i += 32) {
      const float e = __expf(row[i] - m);
      row[i] = e;
      ssum += e;
    }
    #pragma unroll
    for (int off = 16; off >= 1; off >>= 1) ssum += __shfl_xor(ssum, off, 32);
    const float inv = 1.0f / ssum;
    float* dst = attn_out + ((size_t)bh * Sc + (q0 + r)) * Sc;
    for (int i = lane; i < Sc; i += 32) {
      const float p = row[i] * inv;
      row[i] = p;
      dst[i] = p;
    }
  }
  __syncthreads();

  // --- Phase 3: out = P @ V  (P from LDS, V^T from global) --------------
  v8f o[4] = {};
  for (int kt = 0; kt < Sc / 32; ++kt) {
    const float* pr = sc + (wave * 16 + lo) * Sc + kt * 32 + (hi << 4);
    v16h pa = cvt16f(pr);                     // ds_load_b128 x4 + cvt
    #pragma unroll
    for (int nt = 0; nt < 4; ++nt) {
      const int dcol = nt * 16 + lo;
      const _Float16* vrow =
          vpT + ((size_t)bh * Dc + dcol) * Sc + kt * 32 + (hi << 4);
      v16h vb = *(const v16h*)vrow;
      o[nt] = wmma16(pa, vb, o[nt]);
    }
  }

  #pragma unroll
  for (int nt = 0; nt < 4; ++nt) {
    #pragma unroll
    for (int r = 0; r < 8; ++r) {
      const int qr = q0 + r + 8 * hi;
      const int dcol = nt * 16 + lo;
      out_h[(size_t)(b * Sc + qr) * Ec + h * Dc + dcol] = (_Float16)o[nt][r];
    }
  }
}

// ---------------------------------------------------------------------------
extern "C" void kernel_launch(void* const* d_in, const int* in_sizes, int n_in,
                              void* d_out, int out_size, void* d_ws, size_t ws_size,
                              hipStream_t stream) {
  const float* q  = (const float*)d_in[0];
  const float* k  = (const float*)d_in[1];
  const float* v  = (const float*)d_in[2];
  const unsigned char* mask = (const unsigned char*)d_in[3];
  const float* Wq = (const float*)d_in[4];
  const float* bq = (const float*)d_in[5];
  const float* Wk = (const float*)d_in[6];
  const float* bk = (const float*)d_in[7];
  const float* Wv = (const float*)d_in[8];
  const float* bv = (const float*)d_in[9];
  const float* Wo = (const float*)d_in[10];
  const float* bo = (const float*)d_in[11];
  const float* rp = (const float*)d_in[12];

  float* out  = (float*)d_out;                               // [B,S,E]
  float* attn = (float*)d_out + (size_t)Bc * Sc * Ec;        // [B,H,S,S]

  // Workspace carve-up
  char* ws = (char*)d_ws;
  const size_t act_elems = (size_t)Bc * Sc * Ec;             // 8M
  const size_t w_elems   = (size_t)Ec * Ec;                  // 1M
  _Float16* q16  = (_Float16*)ws;  ws += act_elems * 2;      // reused as outh
  _Float16* k16  = (_Float16*)ws;  ws += act_elems * 2;
  _Float16* v16  = (_Float16*)ws;  ws += act_elems * 2;
  _Float16* wq16 = (_Float16*)ws;  ws += w_elems * 2;
  _Float16* wk16 = (_Float16*)ws;  ws += w_elems * 2;
  _Float16* wv16 = (_Float16*)ws;  ws += w_elems * 2;
  _Float16* wo16 = (_Float16*)ws;  ws += w_elems * 2;
  _Float16* rp16 = (_Float16*)ws;  ws += ((NREL * Dc * 2 + 255) / 256) * 256;
  _Float16* qp   = (_Float16*)ws;  ws += act_elems * 2;
  _Float16* kp   = (_Float16*)ws;  ws += act_elems * 2;
  _Float16* vpT  = (_Float16*)ws;  ws += act_elems * 2;
  float*    rel  = (float*)ws;     // B*H*S * 65 f32 (~34 MB)
  _Float16* outh = q16;            // q16 is dead after Q projection

  // 1) f32 -> f16 conversions (activations, weights, rel table)
  const int actN8 = (int)(act_elems / 8), wN8 = (int)(w_elems / 8);
  f32_to_f16_kernel<<<(actN8 + 255) / 256, 256, 0, stream>>>(q, q16, actN8);
  f32_to_f16_kernel<<<(actN8 + 255) / 256, 256, 0, stream>>>(k, k16, actN8);
  f32_to_f16_kernel<<<(actN8 + 255) / 256, 256, 0, stream>>>(v, v16, actN8);
  f32_to_f16_kernel<<<(wN8 + 255) / 256, 256, 0, stream>>>(Wq, wq16, wN8);
  f32_to_f16_kernel<<<(wN8 + 255) / 256, 256, 0, stream>>>(Wk, wk16, wN8);
  f32_to_f16_kernel<<<(wN8 + 255) / 256, 256, 0, stream>>>(Wv, wv16, wN8);
  f32_to_f16_kernel<<<(wN8 + 255) / 256, 256, 0, stream>>>(Wo, wo16, wN8);
  f32_to_f16_kernel<<<(NREL * Dc / 8 + 255) / 256, 256, 0, stream>>>(
      rp, rp16, NREL * Dc / 8);

  // 2) Q/K/V projections (WMMA GEMMs, pure f16 operand stream)
  const dim3 gemm_grid(Bc * Sc / 64, Ec / 64);   // (128, 16)
  gemm_f16_kernel<<<gemm_grid, 128, 0, stream>>>(q16, wq16, bq, qp,  nullptr, 0);
  gemm_f16_kernel<<<gemm_grid, 128, 0, stream>>>(k16, wk16, bk, kp,  nullptr, 1);
  gemm_f16_kernel<<<gemm_grid, 128, 0, stream>>>(v16, wv16, bv, vpT, nullptr, 2);

  // 3) relative-position bias GEMM: [B*H*S] x 65
  rel_scores_kernel<<<(Bc * Hc * Sc) / 64, 128, 0, stream>>>(qp, rp16, rel);

  // 4) fused scores + softmax + weights-out + P@V (async K staging)
  const size_t lds_bytes = 32 * Sc * sizeof(float) + 2 * 2 * 2048;
  attn_kernel<<<dim3(Sc / 32, Bc * Hc), 64, lds_bytes, stream>>>(
      qp, kp, vpT, rel, mask, attn, outh);

  // 5) output projection -> f32 d_out
  gemm_f16_kernel<<<gemm_grid, 128, 0, stream>>>(outh, wo16, bo, nullptr, out, 3);
}
